// MultiheadSelfAttentionEinops_86380382257750
// MI455X (gfx1250) — matrix operand order repaired and checked by reading, not compiled, not run on hardware
//
#include <hip/hip_runtime.h>

#define DEV __device__ __forceinline__

typedef __bf16 v16bf __attribute__((ext_vector_type(16)));
typedef __bf16 v8bf  __attribute__((ext_vector_type(8)));
typedef float  v8f   __attribute__((ext_vector_type(8)));
typedef unsigned int u32x4 __attribute__((ext_vector_type(4)));
typedef int          i32x8 __attribute__((ext_vector_type(8)));
typedef int          i32x4 __attribute__((ext_vector_type(4)));

constexpr int Bn  = 2;
constexpr int Sn  = 2048;
constexpr int Dn  = 1024;
constexpr int Hn  = 16;
constexpr int DHn = 64;
constexpr int BS  = Bn * Sn;   // 4096 rows of x
constexpr int D3  = 3 * Dn;    // 3072 qkv cols

// LDS A-tile layout: 16 rows x 1024 bf16, TDM pads 4 DWORDs after every 256
// data DWORDs (=512 bf16) -> row stride 520 DW = 1040 elems (2-way banked).
constexpr int LDA = 1040;

DEV v8f zero8() { v8f z = {0.f,0.f,0.f,0.f,0.f,0.f,0.f,0.f}; return z; }

DEV v8f wmma_bf16(v16bf a, v16bf b, v8f c) {
  return __builtin_amdgcn_wmma_f32_16x16x32_bf16(
      /*neg_a=*/false, a, /*neg_b=*/false, b,
      /*c_mod=*/(short)0, c, /*reuse_a=*/false, /*reuse_b=*/false);
}

// A-fragment: 16x32 bf16, row-major source with leading dim ld.
// lane: m = lane%16 (row), sub = lane/16 selects K halves {sub*8..+7, 16+sub*8..+7}
DEV v16bf load_fragA(const __bf16* base, int row0, int ld, int k0) {
  int lane = threadIdx.x & 31;
  int m = lane & 15, sub = lane >> 4;
  const __bf16* p = base + (size_t)(row0 + m) * ld + k0 + sub * 8;
  v8bf lo = *(const v8bf*)p;
  v8bf hi = *(const v8bf*)(p + 16);
  v16bf f;
#pragma unroll
  for (int i = 0; i < 8; ++i) { f[i] = lo[i]; f[8 + i] = hi[i]; }
  return f;
}

// A-fragment from the TDM-padded LDS tile (16 x 1024, row stride LDA elems).
DEV v16bf load_fragA_lds(const __bf16* base, int k0) {
  int lane = threadIdx.x & 31;
  int m = lane & 15, sub = lane >> 4;
  int adj = (k0 >= 512) ? 8 : 0;   // one 4-DWORD pad crossed at elem 512
  const __bf16* p = base + m * LDA + adj + k0 + sub * 8;
  v8bf lo = *(const v8bf*)p;
  v8bf hi = *(const v8bf*)(p + 16);
  v16bf f;
#pragma unroll
  for (int i = 0; i < 8; ++i) { f[i] = lo[i]; f[8 + i] = hi[i]; }
  return f;
}

// B-fragment: 32x16 bf16 for C = A x B with B[k][n] = W[n][k]; source W row-major [N,K].
DEV v16bf load_fragB(const __bf16* base, int n0, int ld, int k0) {
  int lane = threadIdx.x & 31;
  int n = lane & 15, sub = lane >> 4;
  const __bf16* p = base + (size_t)(n0 + n) * ld + k0 + sub * 8;
  v8bf lo = *(const v8bf*)p;
  v8bf hi = *(const v8bf*)(p + 16);
  v16bf f;
#pragma unroll
  for (int i = 0; i < 8; ++i) { f[i] = lo[i]; f[8 + i] = hi[i]; }
  return f;
}

// TDM: DMA a 16 x 1024 bf16 tile (row stride Dn) from global into LDS.
// D# per CDNA5 ISA ch.8: group0 = {count=1 | lds_addr | 57b global addr | type=2}
// group1 = {data_size=2B, pad_enable, pad_interval=256DW, pad_amount=4DW,
//           tensor_dim=(1024,16), tile_dim=(1024,16), dim0_stride=1024}
DEV void tdm_load_A16(const __bf16* gsrc, __bf16* ldst) {
  unsigned long long ga = (unsigned long long)(size_t)gsrc;
  u32x4 g0;
  g0[0] = 1u;                                       // count=1, user mode
  g0[1] = (unsigned)(size_t)ldst;                   // LDS byte offset
  g0[2] = (unsigned)ga;                             // global addr [95:64]
  g0[3] = (unsigned)((ga >> 32) & 0x01FFFFFFu) | 0x80000000u;  // addr hi | type=2
  i32x8 g1;
  g1[0] = 0x07D10000;            // data_size=1(2B) | pad_en | interval=7 | amount=3
  g1[1] = (int)(1024u << 16);    // tensor_dim0[15:0] in bits[63:48]
  g1[2] = (int)(16u << 16);      // tensor_dim0 hi=0 | tensor_dim1 lo=16
  g1[3] = (int)(1024u << 16);    // tensor_dim1 hi=0 | tile_dim0=1024
  g1[4] = 16;                    // tile_dim1=16 | tile_dim2=0
  g1[5] = 1024;                  // tensor_dim0_stride lo32
  g1[6] = 0;
  g1[7] = 0;
  i32x4 z4 = {0, 0, 0, 0};
#if defined(__clang_major__) && __clang_major__ >= 23
  i32x8 z8 = {0, 0, 0, 0, 0, 0, 0, 0};
  __builtin_amdgcn_tensor_load_to_lds(g0, g1, z4, z4, z8, 0);
#else
  __builtin_amdgcn_tensor_load_to_lds(g0, g1, z4, z4, 0);
#endif
}

// ---------------- stage 0: f32 -> bf16 ----------------
__global__ void cvt_f32_bf16(const float* __restrict__ in, __bf16* __restrict__ out, int n) {
  int i = blockIdx.x * blockDim.x + threadIdx.x;
  int stride = gridDim.x * blockDim.x;
  for (; i < n; i += stride) out[i] = (__bf16)in[i];
}

// ---------------- stage 1: QKV GEMM + bias + RoPE, scatter to Q/K/Vt ----------------
// grid: (BS/16, D3/64/8), block 256 (8 waves). Shared 16-row A tile staged by TDM.
__global__ __launch_bounds__(256) void qkv_rope_kernel(
    const __bf16* __restrict__ xb, const __bf16* __restrict__ wb,
    const float* __restrict__ bias,
    __bf16* __restrict__ Qb, __bf16* __restrict__ Kb, __bf16* __restrict__ Vt) {
  __shared__ __attribute__((aligned(16))) __bf16 ldsA[16 * LDA];
  const int wave = threadIdx.x >> 5, lane = threadIdx.x & 31;
  const int sub = lane >> 4, ln = lane & 15;
  const int row0 = blockIdx.x * 16;
  const int n0 = (blockIdx.y * 8 + wave) * 64;

  if (threadIdx.x < 32) {                       // wave 0 drives the TDM
    tdm_load_A16(xb + (size_t)row0 * Dn, ldsA);
    __builtin_amdgcn_s_wait_tensorcnt(0);
  }
  __syncthreads();

  v8f acc[4];
#pragma unroll
  for (int j = 0; j < 4; ++j) acc[j] = zero8();

  for (int kk = 0; kk < Dn; kk += 32) {
    v16bf a = load_fragA_lds(ldsA, kk);
#pragma unroll
    for (int j = 0; j < 4; ++j)
      acc[j] = wmma_bf16(a, load_fragB(wb, n0 + j * 16, Dn, kk), acc[j]);
  }

  const int sel = n0 / Dn;            // 0=q 1=k 2=v
  const int h   = (n0 % Dn) / DHn;

  float vals[4][8];
#pragma unroll
  for (int j = 0; j < 4; ++j) {
    float bj = bias[n0 + j * 16 + ln];
#pragma unroll
    for (int r = 0; r < 8; ++r) vals[j][r] = acc[j][r] + bj;
  }

  const float L2_BASE = 13.287712379549449f;  // log2(10000)
#pragma unroll
  for (int j = 0; j < 4; ++j) {
    const int dh = j * 16 + ln;
    const float invf = exp2f(-(float)(dh & 31) * (L2_BASE / 32.0f));
    const int pj = j ^ 2;  // rotate_half partner tile
#pragma unroll
    for (int r = 0; r < 8; ++r) {
      const int rowg = row0 + r + 8 * sub;
      const int b = rowg >> 11;        // / Sn
      const int s = rowg & (Sn - 1);
      if (sel == 2) {
        Vt[(((size_t)(b * Hn + h)) * DHn + dh) * Sn + s] = (__bf16)vals[j][r];
      } else {
        float ang = (float)s * invf;
        float c = __cosf(ang), sn = __sinf(ang);
        float rot = (j < 2) ? -vals[pj][r] : vals[pj][r];
        float o = vals[j][r] * c + rot * sn;
        size_t idx = (((size_t)(b * Hn + h)) * Sn + s) * DHn + dh;
        if (sel == 0) Qb[idx] = (__bf16)(o * 0.125f);  // fold 1/sqrt(Dh)
        else          Kb[idx] = (__bf16)o;
      }
    }
  }
}

// ---------------- stage 2: flash attention ----------------
// grid: B*H*(S/64), block 128 (4 waves). Wave = 16 query rows, streams 32-key blocks.
__global__ __launch_bounds__(128) void flash_attn_kernel(
    const __bf16* __restrict__ Qb, const __bf16* __restrict__ Kb,
    const __bf16* __restrict__ Vt, __bf16* __restrict__ ctxb) {
  __shared__ __bf16 ldsP[4][16][32];
  const int wave = threadIdx.x >> 5, lane = threadIdx.x & 31;
  const int sub = lane >> 4, ln = lane & 15;
  const int RB = Sn / 64;
  const int bh = blockIdx.x / RB, rb = blockIdx.x % RB;
  const int b = bh / Hn, h = bh % Hn;
  const __bf16* Qp = Qb + (size_t)bh * Sn * DHn;
  const __bf16* Kp = Kb + (size_t)bh * Sn * DHn;
  const __bf16* Vp = Vt + (size_t)bh * DHn * Sn;
  const int qbase = rb * 64 + wave * 16;

  v16bf aq0 = load_fragA(Qp, qbase, DHn, 0);
  v16bf aq1 = load_fragA(Qp, qbase, DHn, 32);

  float m[8], l[8];
  v8f cacc[4];
#pragma unroll
  for (int r = 0; r < 8; ++r) { m[r] = -3.0e38f; l[r] = 0.f; }
#pragma unroll
  for (int j = 0; j < 4; ++j) cacc[j] = zero8();

  const int qmax = qbase + 15;
  for (int t0 = 0; t0 <= qmax; t0 += 32) {
    v8f s0 = zero8(), s1 = zero8();
    s0 = wmma_bf16(aq0, load_fragB(Kp, t0, DHn, 0), s0);
    s0 = wmma_bf16(aq1, load_fragB(Kp, t0, DHn, 32), s0);
    s1 = wmma_bf16(aq0, load_fragB(Kp, t0 + 16, DHn, 0), s1);
    s1 = wmma_bf16(aq1, load_fragB(Kp, t0 + 16, DHn, 32), s1);

    const int c0 = t0 + ln, c1 = c0 + 16;
#pragma unroll
    for (int r = 0; r < 8; ++r) {
      const int q = qbase + r + 8 * sub;
      float e0 = (c0 <= q) ? s0[r] : -3.0e38f;  // causal mask
      float e1 = (c1 <= q) ? s1[r] : -3.0e38f;
      float loc = fmaxf(e0, e1);
      loc = fmaxf(loc, __shfl_xor(loc, 1, 16));
      loc = fmaxf(loc, __shfl_xor(loc, 2, 16));
      loc = fmaxf(loc, __shfl_xor(loc, 4, 16));
      loc = fmaxf(loc, __shfl_xor(loc, 8, 16));
      float mn = fmaxf(m[r], loc);
      float corr = __expf(m[r] - mn);
      m[r] = mn;
      float p0 = __expf(e0 - mn);
      float p1 = __expf(e1 - mn);
      float rs = p0 + p1;
      rs += __shfl_xor(rs, 1, 16);
      rs += __shfl_xor(rs, 2, 16);
      rs += __shfl_xor(rs, 4, 16);
      rs += __shfl_xor(rs, 8, 16);
      l[r] = l[r] * corr + rs;
#pragma unroll
      for (int j = 0; j < 4; ++j) cacc[j][r] *= corr;
      const int row = r + 8 * sub;
      ldsP[wave][row][ln]      = (__bf16)p0;   // C-layout -> row-major P tile
      ldsP[wave][row][16 + ln] = (__bf16)p1;
    }
    // P (16x32) as A-fragment from LDS (DS ops are in-order per wave)
    v16bf pa = load_fragA(&ldsP[wave][0][0], 0, 32, 0);
#pragma unroll
    for (int j = 0; j < 4; ++j)
      cacc[j] = wmma_bf16(pa, load_fragB(Vp, j * 16, Sn, t0), cacc[j]);
  }

#pragma unroll
  for (int r = 0; r < 8; ++r) {
    const float inv = 1.0f / l[r];
    const int q = qbase + r + 8 * sub;
    const size_t off = ((size_t)(b * Sn + q)) * Dn + h * DHn;
#pragma unroll
    for (int j = 0; j < 4; ++j)
      ctxb[off + j * 16 + ln] = (__bf16)(cacc[j][r] * inv);
  }
}

// ---------------- stage 3: output projection ----------------
// grid: (BS/16, Dn/64/8), block 256. Shared A tile staged by TDM.
__global__ __launch_bounds__(256) void proj_kernel(
    const __bf16* __restrict__ ctxb, const __bf16* __restrict__ wb,
    const float* __restrict__ bias, float* __restrict__ out) {
  __shared__ __attribute__((aligned(16))) __bf16 ldsA[16 * LDA];
  const int wave = threadIdx.x >> 5, lane = threadIdx.x & 31;
  const int sub = lane >> 4, ln = lane & 15;
  const int row0 = blockIdx.x * 16;
  const int n0 = (blockIdx.y * 8 + wave) * 64;

  if (threadIdx.x < 32) {
    tdm_load_A16(ctxb + (size_t)row0 * Dn, ldsA);
    __builtin_amdgcn_s_wait_tensorcnt(0);
  }
  __syncthreads();

  v8f acc[4];
#pragma unroll
  for (int j = 0; j < 4; ++j) acc[j] = zero8();

  for (int kk = 0; kk < Dn; kk += 32) {
    v16bf a = load_fragA_lds(ldsA, kk);
#pragma unroll
    for (int j = 0; j < 4; ++j)
      acc[j] = wmma_bf16(a, load_fragB(wb, n0 + j * 16, Dn, kk), acc[j]);
  }

#pragma unroll
  for (int j = 0; j < 4; ++j) {
    float bj = bias[n0 + j * 16 + ln];
#pragma unroll
    for (int r = 0; r < 8; ++r) {
      int rowg = row0 + r + 8 * sub;
      out[(size_t)rowg * Dn + n0 + j * 16 + ln] = acc[j][r] + bj;
    }
  }
}

extern "C" void kernel_launch(void* const* d_in, const int* in_sizes, int n_in,
                              void* d_out, int out_size, void* d_ws, size_t ws_size,
                              hipStream_t stream) {
  (void)in_sizes; (void)n_in; (void)out_size; (void)ws_size;
  const float* x      = (const float*)d_in[0];
  const float* qkv_w  = (const float*)d_in[1];
  const float* qkv_b  = (const float*)d_in[2];
  const float* proj_w = (const float*)d_in[3];
  const float* proj_b = (const float*)d_in[4];
  float* out = (float*)d_out;

  char* ws = (char*)d_ws;
  size_t o = 0;
  auto alloc = [&](size_t nelem) -> __bf16* {
    __bf16* p = (__bf16*)(ws + o);
    o += nelem * sizeof(__bf16);
    return p;
  };
  __bf16* xb    = alloc((size_t)BS * Dn);        // x in bf16
  __bf16* wqkv  = alloc((size_t)D3 * Dn);        // qkv_w in bf16
  __bf16* wproj = alloc((size_t)Dn * Dn);        // proj_w in bf16
  __bf16* Qb    = alloc((size_t)BS * Dn);        // [B,H,S,Dh] rope'd, scaled
  __bf16* Kb    = alloc((size_t)BS * Dn);        // [B,H,S,Dh] rope'd
  __bf16* Vt    = alloc((size_t)BS * Dn);        // [B,H,Dh,S] transposed
  __bf16* ctxb  = alloc((size_t)BS * Dn);        // [B,S,D]

  cvt_f32_bf16<<<2048, 256, 0, stream>>>(x, xb, BS * Dn);
  cvt_f32_bf16<<<2048, 256, 0, stream>>>(qkv_w, wqkv, D3 * Dn);
  cvt_f32_bf16<<<1024, 256, 0, stream>>>(proj_w, wproj, Dn * Dn);

  qkv_rope_kernel<<<dim3(BS / 16, D3 / 64 / 8), 256, 0, stream>>>(
      xb, wqkv, qkv_b, Qb, Kb, Vt);

  flash_attn_kernel<<<Bn * Hn * (Sn / 64), 128, 0, stream>>>(Qb, Kb, Vt, ctxb);

  proj_kernel<<<dim3(BS / 16, Dn / 64 / 8), 256, 0, stream>>>(
      ctxb, wproj, proj_b, out);
}
